// LegalMemoryAugmentedNetwork_867583393906
// MI455X (gfx1250) — compile-verified, matrix operand order checked by reading, not run on hardware
//
#include <hip/hip_runtime.h>
#include <math.h>

// ---------------------------------------------------------------------------
// Problem constants
// ---------------------------------------------------------------------------
#define EE 768   // embed
#define HH 512   // hidden
#define FH 2048  // 4*H
#define MM_ 512  // memory slots
#define DD 768   // memory vector dim
#define BB 8     // batch
#define TT 512   // seq len
#define NT (BB * TT)  // 4096 flattened rows (row = b*T + t)

typedef __bf16 bf16x16 __attribute__((ext_vector_type(16)));
typedef float  floatx8 __attribute__((ext_vector_type(8)));
typedef unsigned short u16;

__device__ __forceinline__ u16 f32_to_bf16_bits(float f) {
    union { float f; unsigned u; } v; v.f = f;
    unsigned r = v.u + 0x7FFFu + ((v.u >> 16) & 1u);  // round-to-nearest-even
    return (u16)(r >> 16);
}
__device__ __forceinline__ float sigf(float x) { return 1.0f / (1.0f + __expf(-x)); }

// ---------------------------------------------------------------------------
// f32 -> bf16 conversion
// ---------------------------------------------------------------------------
__global__ void cvt_bf16_kernel(const float* __restrict__ in, u16* __restrict__ out, size_t n) {
    size_t i = (size_t)blockIdx.x * blockDim.x + threadIdx.x;
    if (i < n) out[i] = f32_to_bf16_bits(in[i]);
}

// ---------------------------------------------------------------------------
// Generic WMMA GEMM:  C[n,p] = sum_k A[n,k]*W[p,k] + bias[p]   (optional sigmoid)
// A: [N,K] bf16 row-major, W: [P,K] bf16 row-major (K contiguous for both).
// One wave per 16x16 output tile; K multiple of 32.
// A-frag: lane l holds row (l&15), K chunk (l>>4)*16..+16  (contiguous k = 16h+s order)
// B-frag: lane l holds col (l&15) = W row, same K chunk     (same contiguous k order as A)
// C-frag: VGPR r, lane l -> row r + 8*(l>>4), col (l&15)
// ---------------------------------------------------------------------------
__global__ __launch_bounds__(32) void wmma_gemm_bias(
    const u16* __restrict__ A, const u16* __restrict__ W,
    const float* __restrict__ bias, float* __restrict__ C,
    int K, int P, int act)
{
    const int lane = threadIdx.x;
    const int ln = lane & 15, lh = lane >> 4;
    const int arow = blockIdx.y * 16 + ln;
    const int col  = blockIdx.x * 16 + ln;
    const u16* a_ptr = A + (size_t)arow * K + lh * 16;
    const u16* w_ptr = W + (size_t)col  * K + lh * 16;

    floatx8 acc;
    float bv = bias ? bias[col] : 0.0f;
    #pragma unroll
    for (int r = 0; r < 8; ++r) acc[r] = bv;

    for (int k = 0; k < K; k += 32) {
        bf16x16 a = *(const bf16x16*)(a_ptr + k);
        bf16x16 b = *(const bf16x16*)(w_ptr + k);
        acc = __builtin_amdgcn_wmma_f32_16x16x32_bf16(false, a, false, b,
                                                      (short)0, acc, false, false);
    }
    #pragma unroll
    for (int r = 0; r < 8; ++r) {
        int mrow = blockIdx.y * 16 + r + 8 * lh;
        float v = acc[r];
        if (act == 1) v = sigf(v);
        C[(size_t)mrow * P + blockIdx.x * 16 + ln] = v;
    }
}

// ---------------------------------------------------------------------------
// Persistent single-workgroup LSTM layer.
// G: [B*T, 4H] f32 = x@Wih^T + bih   (rows b*T+t)
// Whh: [4H, H] bf16, bhh: [4H]
// 32 waves; wave w owns hidden columns [16w,16w+16). Cell state c lives in
// registers for all T steps; h double-buffered in LDS (bf16), one s_barrier/step.
// ---------------------------------------------------------------------------
__global__ __launch_bounds__(1024) void lstm_layer_wmma(
    const float* __restrict__ G, const u16* __restrict__ Whh,
    const float* __restrict__ bhh,
    u16* __restrict__ hseq_bf,          // [B*T, H] bf16 (rows b*T+t)
    float* __restrict__ hseq_f32)       // optional same layout (nullptr ok)
{
    __shared__ __align__(32) u16 hbuf[2][16][HH];
    const int tid = threadIdx.x;
    for (int i = tid; i < 2 * 16 * HH; i += 1024) ((u16*)hbuf)[i] = 0;
    __syncthreads();

    const int wave = tid >> 5, lane = tid & 31;
    const int ln = lane & 15, lh = lane >> 4;
    const int jcol = wave * 16 + ln;          // hidden column (B-frag / C col)

    const float b_i = bhh[0 * HH + jcol];
    const float b_f = bhh[1 * HH + jcol];
    const float b_g = bhh[2 * HH + jcol];
    const float b_o = bhh[3 * HH + jcol];

    const u16* w_i = Whh + (size_t)(0 * HH + jcol) * HH + lh * 16;
    const u16* w_f = Whh + (size_t)(1 * HH + jcol) * HH + lh * 16;
    const u16* w_g = Whh + (size_t)(2 * HH + jcol) * HH + lh * 16;
    const u16* w_o = Whh + (size_t)(3 * HH + jcol) * HH + lh * 16;

    float c[8];
    #pragma unroll
    for (int r = 0; r < 8; ++r) c[r] = 0.0f;

    int p = 0;
    for (int t = 0; t < TT; ++t) {
        floatx8 a_i, a_f, a_g, a_o;
        #pragma unroll
        for (int r = 0; r < 8; ++r) {
            int mmr = r + 8 * lh;             // C-frag row (batch; >=8 is pad)
            float gi = 0.f, gf = 0.f, gg = 0.f, go = 0.f;
            if (mmr < BB) {
                const float* gp = G + ((size_t)(mmr * TT + t)) * FH;
                gi = gp[0 * HH + jcol] + b_i;
                gf = gp[1 * HH + jcol] + b_f;
                gg = gp[2 * HH + jcol] + b_g;
                go = gp[3 * HH + jcol] + b_o;
            }
            a_i[r] = gi; a_f[r] = gf; a_g[r] = gg; a_o[r] = go;
        }
        const u16* hb = &hbuf[p][ln][lh * 16];
        for (int k = 0; k < HH; k += 32) {
            bf16x16 a  = *(const bf16x16*)(hb + k);
            bf16x16 wi = *(const bf16x16*)(w_i + k);
            bf16x16 wf = *(const bf16x16*)(w_f + k);
            bf16x16 wg = *(const bf16x16*)(w_g + k);
            bf16x16 wo = *(const bf16x16*)(w_o + k);
            a_i = __builtin_amdgcn_wmma_f32_16x16x32_bf16(false, a, false, wi, (short)0, a_i, false, false);
            a_f = __builtin_amdgcn_wmma_f32_16x16x32_bf16(false, a, false, wf, (short)0, a_f, false, false);
            a_g = __builtin_amdgcn_wmma_f32_16x16x32_bf16(false, a, false, wg, (short)0, a_g, false, false);
            a_o = __builtin_amdgcn_wmma_f32_16x16x32_bf16(false, a, false, wo, (short)0, a_o, false, false);
        }
        const int pn = p ^ 1;
        #pragma unroll
        for (int r = 0; r < 8; ++r) {
            int mmr = r + 8 * lh;
            if (mmr < BB) {
                float i_ = sigf(a_i[r]);
                float f_ = sigf(a_f[r]);
                float g_ = tanhf(a_g[r]);
                float o_ = sigf(a_o[r]);
                float cc = f_ * c[r] + i_ * g_;
                c[r] = cc;
                float h = o_ * tanhf(cc);
                u16 hb16 = f32_to_bf16_bits(h);
                hbuf[pn][mmr][jcol] = hb16;
                size_t off = ((size_t)(mmr * TT + t)) * HH + jcol;
                hseq_bf[off] = hb16;
                if (hseq_f32) hseq_f32[off] = h;
            }
        }
        __syncthreads();
        p = pn;
    }
}

// ---------------------------------------------------------------------------
// In-place row softmax (rows of length `cols`), one block per row.
// ---------------------------------------------------------------------------
__global__ __launch_bounds__(256) void softmax_rows(float* __restrict__ X, int cols) {
    float* x = X + (size_t)blockIdx.x * cols;
    __shared__ float red[256];
    const int tid = threadIdx.x;
    float mx = -1e30f;
    for (int i = tid; i < cols; i += 256) mx = fmaxf(mx, x[i]);
    red[tid] = mx; __syncthreads();
    for (int s = 128; s > 0; s >>= 1) { if (tid < s) red[tid] = fmaxf(red[tid], red[tid + s]); __syncthreads(); }
    mx = red[0]; __syncthreads();
    float sum = 0.f;
    for (int i = tid; i < cols; i += 256) { float e = __expf(x[i] - mx); x[i] = e; sum += e; }
    red[tid] = sum; __syncthreads();
    for (int s = 128; s > 0; s >>= 1) { if (tid < s) red[tid] += red[tid + s]; __syncthreads(); }
    float inv = 1.0f / red[0];
    for (int i = tid; i < cols; i += 256) x[i] *= inv;
}

// ---------------------------------------------------------------------------
// Sequential memory scan. Columns are independent except for the read GEMV,
// so: 48 workgroups x 16 d-columns; each thread keeps 32 memory elements in
// REGISTERS across all 512 steps. Per step: stage rw/ew/ww rows in LDS,
// compute reads via LDS tree reduction, then elementwise (t,b)-ordered update,
// streaming memory_states (the 805MB store-bound phase) with 64B-contiguous rows.
// ---------------------------------------------------------------------------
__global__ __launch_bounds__(256) void memory_scan(
    const float* __restrict__ mem0,     // [M, D]
    const float* __restrict__ rw,       // [B*T, M] (rows b*T+t), softmaxed
    const float* __restrict__ wwg,      // write_w
    const float* __restrict__ ewg,      // erase_w (sigmoid)
    const float* __restrict__ wv,       // [B*T, D]
    const float* __restrict__ ev,       // sigmoid
    float* __restrict__ reads,          // [B*T, D]
    float* __restrict__ states,         // [T, M, D]
    float* __restrict__ finalmem)       // [M, D]
{
    __shared__ float rw_s[BB * MM_];
    __shared__ float ew_s[BB * MM_];
    __shared__ float ww_s[BB * MM_];
    __shared__ float red[BB * 16 * 16];

    const int tid = threadIdx.x;
    const int dl = tid & 15, mg = tid >> 4;   // 16 d-lanes x 16 m-groups
    const int d  = blockIdx.x * 16 + dl;
    const int m0 = mg * 32;

    float v[32];
    #pragma unroll
    for (int i = 0; i < 32; ++i) v[i] = mem0[(size_t)(m0 + i) * DD + d];

    for (int t = 0; t < TT; ++t) {
        for (int idx = tid; idx < BB * MM_; idx += 256) {
            int b = idx >> 9, m = idx & (MM_ - 1);
            size_t goff = ((size_t)(b * TT + t)) * MM_ + m;
            rw_s[idx] = rw[goff];
            ew_s[idx] = ewg[goff];
            ww_s[idx] = wwg[goff];
        }
        __syncthreads();

        // read BEFORE this step's writes
        #pragma unroll
        for (int b = 0; b < BB; ++b) {
            const float* r = rw_s + b * MM_ + m0;
            float acc = 0.f;
            #pragma unroll
            for (int i = 0; i < 32; ++i) acc += r[i] * v[i];
            red[(b * 16 + mg) * 16 + dl] = acc;
        }
        __syncthreads();
        if (tid < 128) {
            int b = tid >> 4, dd = tid & 15;
            float s = 0.f;
            #pragma unroll
            for (int g = 0; g < 16; ++g) s += red[(b * 16 + g) * 16 + dd];
            reads[((size_t)(b * TT + t)) * DD + blockIdx.x * 16 + dd] = s;
        }

        // sequential batch updates (composition order matches reference scan)
        float evv[BB], wvv[BB];
        #pragma unroll
        for (int b = 0; b < BB; ++b) {
            size_t voff = ((size_t)(b * TT + t)) * DD + d;
            evv[b] = ev[voff];
            wvv[b] = wv[voff];
        }
        #pragma unroll
        for (int i = 0; i < 32; ++i) {
            float x = v[i];
            const int m = m0 + i;
            #pragma unroll
            for (int b = 0; b < BB; ++b)
                x = x * (1.0f - ew_s[b * MM_ + m] * evv[b]) + ww_s[b * MM_ + m] * wvv[b];
            v[i] = x;
            states[((size_t)t * MM_ + m) * DD + d] = x;
        }
        __syncthreads();
    }
    #pragma unroll
    for (int i = 0; i < 32; ++i) finalmem[(size_t)(m0 + i) * DD + d] = v[i];
}

// ---------------------------------------------------------------------------
// combined[b, f] = mean_t concat(ctrl, reads)[b, t, f],  f in [0,1280)
// ---------------------------------------------------------------------------
__global__ void mean_combined(const float* __restrict__ ctrl,
                              const float* __restrict__ reads,
                              float* __restrict__ combined)
{
    int idx = blockIdx.x * blockDim.x + threadIdx.x;
    if (idx >= BB * (HH + DD)) return;
    int b = idx / (HH + DD), f = idx % (HH + DD);
    float s = 0.f;
    if (f < HH) {
        const float* p = ctrl + ((size_t)b * TT) * HH + f;
        for (int t = 0; t < TT; ++t) s += p[(size_t)t * HH];
    } else {
        const float* p = reads + ((size_t)b * TT) * DD + (f - HH);
        for (int t = 0; t < TT; ++t) s += p[(size_t)t * DD];
    }
    combined[idx] = s * (1.0f / (float)TT);
}

// ---------------------------------------------------------------------------
// Tiny classifier head: relu(combined@Wc1^T+bc1)@Wc2^T+bc2 -> softmax(3)
// ---------------------------------------------------------------------------
__global__ __launch_bounds__(256) void classifier_kernel(
    const float* __restrict__ combined, const float* __restrict__ Wc1,
    const float* __restrict__ bc1, const float* __restrict__ Wc2,
    const float* __restrict__ bc2, float* __restrict__ probs)
{
    __shared__ float hid[BB * HH];
    __shared__ float logits[BB * 3];
    const int tid = threadIdx.x;
    for (int idx = tid; idx < BB * HH; idx += 256) {
        int b = idx >> 9, j = idx & (HH - 1);
        const float* cb = combined + b * (HH + DD);
        const float* w  = Wc1 + (size_t)j * (HH + DD);
        float s = bc1[j];
        for (int k = 0; k < HH + DD; ++k) s += cb[k] * w[k];
        hid[idx] = fmaxf(s, 0.0f);
    }
    __syncthreads();
    if (tid < BB * 3) {
        int b = tid / 3, k = tid % 3;
        const float* w = Wc2 + k * HH;
        float s = bc2[k];
        for (int j = 0; j < HH; ++j) s += hid[b * HH + j] * w[j];
        logits[tid] = s;
    }
    __syncthreads();
    if (tid < BB) {
        float l0 = logits[tid * 3], l1 = logits[tid * 3 + 1], l2 = logits[tid * 3 + 2];
        float mx = fmaxf(l0, fmaxf(l1, l2));
        float e0 = __expf(l0 - mx), e1 = __expf(l1 - mx), e2 = __expf(l2 - mx);
        float inv = 1.0f / (e0 + e1 + e2);
        probs[tid * 3 + 0] = e0 * inv;
        probs[tid * 3 + 1] = e1 * inv;
        probs[tid * 3 + 2] = e2 * inv;
    }
}

// ---------------------------------------------------------------------------
// Host-side orchestration
// ---------------------------------------------------------------------------
extern "C" void kernel_launch(void* const* d_in, const int* in_sizes, int n_in,
                              void* d_out, int out_size, void* d_ws, size_t ws_size,
                              hipStream_t stream) {
    const float* x      = (const float*)d_in[0];
    const float* memory = (const float*)d_in[1];
    const float* Wih0 = (const float*)d_in[2];
    const float* Whh0 = (const float*)d_in[3];
    const float* bih0 = (const float*)d_in[4];
    const float* bhh0 = (const float*)d_in[5];
    const float* Wih1 = (const float*)d_in[6];
    const float* Whh1 = (const float*)d_in[7];
    const float* bih1 = (const float*)d_in[8];
    const float* bhh1 = (const float*)d_in[9];
    const float* Wr  = (const float*)d_in[10];
    const float* br  = (const float*)d_in[11];
    const float* Ww  = (const float*)d_in[12];
    const float* bw  = (const float*)d_in[13];
    const float* We  = (const float*)d_in[14];
    const float* be  = (const float*)d_in[15];
    const float* Wwv = (const float*)d_in[16];
    const float* bwv = (const float*)d_in[17];
    const float* Wev = (const float*)d_in[18];
    const float* bev = (const float*)d_in[19];
    const float* Wc1 = (const float*)d_in[20];
    const float* bc1 = (const float*)d_in[21];
    const float* Wc2 = (const float*)d_in[22];
    const float* bc2 = (const float*)d_in[23];

    float* out = (float*)d_out;
    float* ctrl_out   = out;                                  // [B,T,H]
    float* reads_out  = ctrl_out + (size_t)BB * TT * HH;      // [B,T,D]
    float* states_out = reads_out + (size_t)BB * TT * DD;     // [T,M,D]
    float* probs_out  = states_out + (size_t)TT * MM_ * DD;   // [B,3]
    float* final_out  = probs_out + BB * 3;                   // [M,D]

    // bump allocator on workspace
    char* ws = (char*)d_ws;
    size_t off = 0;
    auto walloc = [&](size_t bytes) -> void* {
        off = (off + 255) & ~(size_t)255;
        void* p = ws + off;
        off += bytes;
        return p;
    };
    u16* xbf     = (u16*)walloc((size_t)NT * EE * 2);
    u16* wih0bf  = (u16*)walloc((size_t)FH * EE * 2);
    u16* whh0bf  = (u16*)walloc((size_t)FH * HH * 2);
    u16* wih1bf  = (u16*)walloc((size_t)FH * HH * 2);
    u16* whh1bf  = (u16*)walloc((size_t)FH * HH * 2);
    u16* wrbf    = (u16*)walloc((size_t)MM_ * HH * 2);
    u16* wwbf    = (u16*)walloc((size_t)MM_ * HH * 2);
    u16* webf    = (u16*)walloc((size_t)MM_ * HH * 2);
    u16* wwvbf   = (u16*)walloc((size_t)DD * HH * 2);
    u16* wevbf   = (u16*)walloc((size_t)DD * HH * 2);
    u16* h1bf    = (u16*)walloc((size_t)NT * HH * 2);
    u16* ctrlbf  = (u16*)walloc((size_t)NT * HH * 2);
    float* G       = (float*)walloc((size_t)NT * FH * 4);
    float* read_w  = (float*)walloc((size_t)NT * MM_ * 4);
    float* write_w = (float*)walloc((size_t)NT * MM_ * 4);
    float* erase_w = (float*)walloc((size_t)NT * MM_ * 4);
    float* write_v = (float*)walloc((size_t)NT * DD * 4);
    float* erase_v = (float*)walloc((size_t)NT * DD * 4);
    float* combined = (float*)walloc((size_t)BB * (HH + DD) * 4);

    auto cvt = [&](const float* src, u16* dst, size_t n) {
        cvt_bf16_kernel<<<(unsigned)((n + 255) / 256), 256, 0, stream>>>(src, dst, n);
    };
    cvt(x,    xbf,    (size_t)NT * EE);
    cvt(Wih0, wih0bf, (size_t)FH * EE);
    cvt(Whh0, whh0bf, (size_t)FH * HH);
    cvt(Wih1, wih1bf, (size_t)FH * HH);
    cvt(Whh1, whh1bf, (size_t)FH * HH);
    cvt(Wr,   wrbf,   (size_t)MM_ * HH);
    cvt(Ww,   wwbf,   (size_t)MM_ * HH);
    cvt(We,   webf,   (size_t)MM_ * HH);
    cvt(Wwv,  wwvbf,  (size_t)DD * HH);
    cvt(Wev,  wevbf,  (size_t)DD * HH);

    // Layer 0: precompute x@Wih0^T + bih0, then persistent recurrence
    wmma_gemm_bias<<<dim3(FH / 16, NT / 16), 32, 0, stream>>>(xbf, wih0bf, bih0, G, EE, FH, 0);
    lstm_layer_wmma<<<1, 1024, 0, stream>>>(G, whh0bf, bhh0, h1bf, nullptr);

    // Layer 1
    wmma_gemm_bias<<<dim3(FH / 16, NT / 16), 32, 0, stream>>>(h1bf, wih1bf, bih1, G, HH, FH, 0);
    lstm_layer_wmma<<<1, 1024, 0, stream>>>(G, whh1bf, bhh1, ctrlbf, ctrl_out);

    // Heads (A = ctrl bf16, K = H)
    wmma_gemm_bias<<<dim3(MM_ / 16, NT / 16), 32, 0, stream>>>(ctrlbf, wrbf, br, read_w, HH, MM_, 0);
    softmax_rows<<<NT, 256, 0, stream>>>(read_w, MM_);
    wmma_gemm_bias<<<dim3(MM_ / 16, NT / 16), 32, 0, stream>>>(ctrlbf, wwbf, bw, write_w, HH, MM_, 0);
    softmax_rows<<<NT, 256, 0, stream>>>(write_w, MM_);
    wmma_gemm_bias<<<dim3(MM_ / 16, NT / 16), 32, 0, stream>>>(ctrlbf, webf, be, erase_w, HH, MM_, 1);
    wmma_gemm_bias<<<dim3(DD / 16, NT / 16), 32, 0, stream>>>(ctrlbf, wwvbf, bwv, write_v, HH, DD, 0);
    wmma_gemm_bias<<<dim3(DD / 16, NT / 16), 32, 0, stream>>>(ctrlbf, wevbf, bev, erase_v, HH, DD, 1);

    // Sequential memory scan (store-bandwidth bound: streams [T,M,D])
    memory_scan<<<DD / 16, 256, 0, stream>>>(memory, read_w, write_w, erase_w,
                                             write_v, erase_v,
                                             reads_out, states_out, final_out);

    // Classifier
    mean_combined<<<(BB * (HH + DD) + 255) / 256, 256, 0, stream>>>(ctrl_out, reads_out, combined);
    classifier_kernel<<<1, 256, 0, stream>>>(combined, Wc1, bc1, Wc2, bc2, probs_out);
}
